// GNNBranch_47674136986069
// MI455X (gfx1250) — matrix operand order repaired and checked by reading, not compiled, not run on hardware
//
#include <hip/hip_runtime.h>
#include <hip/hip_bf16.h>

typedef __attribute__((ext_vector_type(16))) __bf16 v16bf;
typedef __attribute__((ext_vector_type(8)))  __bf16 v8bf;
typedef __attribute__((ext_vector_type(8)))  float  v8f;

#define NF 64
#define NE 64
#define EF 32
#define EE 64
#define OF 128
#define KF 128  /* EE + NE */

__device__ __forceinline__ float lrelu(float v) {
    return v >= 0.0f ? v : 0.01f * v;
}

#define SHUF16(a, b) __builtin_shufflevector(a, b, 0,1,2,3,4,5,6,7,8,9,10,11,12,13,14,15)

// ---------------------------------------------------------------------------
// Prep: transpose weights and convert to bf16 (WT[n][k] = W[k][n]).
// ---------------------------------------------------------------------------
__global__ void prep_weights(const float* __restrict__ Wn, const float* __restrict__ We,
                             const float* __restrict__ Wg,
                             __bf16* __restrict__ WnT, __bf16* __restrict__ WeT,
                             __bf16* __restrict__ WgT) {
    int tid = blockIdx.x * blockDim.x + threadIdx.x;
    int stride = gridDim.x * blockDim.x;
    for (int i = tid; i < NF * NE; i += stride) {
        int k = i / NE, n = i % NE;
        WnT[n * NF + k] = (__bf16)Wn[i];
    }
    for (int i = tid; i < EF * EE; i += stride) {
        int k = i / EE, n = i % EE;
        WeT[n * EF + k] = (__bf16)We[i];
    }
    for (int i = tid; i < KF * OF; i += stride) {
        int k = i / OF, n = i % OF;
        WgT[n * KF + k] = (__bf16)Wg[i];
    }
}

// ---------------------------------------------------------------------------
// Node encoder: node_enc = lrelu(x @ Wn + bn), stored bf16 [N][64].
// One wave per 16-node tile; bias folded into WMMA accumulator init.
// `tile` forced into an SGPR so tile-level guards are scalar branches.
// ---------------------------------------------------------------------------
__global__ void __launch_bounds__(256) node_encoder(
    const float* __restrict__ x, const __bf16* __restrict__ WnT,
    const float* __restrict__ bn, __bf16* __restrict__ node_enc,
    int N, int nTiles)
{
    const int lane = threadIdx.x & 31;
    const int wave = threadIdx.x >> 5;
    const int lo = lane & 15;
    const int hi = lane >> 4;
    const int kh8 = 8 * hi;

    // Wave-uniform tile index in an SGPR -> scalar control flow.
    int tile = __builtin_amdgcn_readfirstlane(blockIdx.x * (blockDim.x >> 5) + wave);
    const int tstride = gridDim.x * (blockDim.x >> 5);
    for (; tile < nTiles; tile += tstride) {
        int rowA = tile * 16 + lo;
        if (rowA >= N) rowA = N - 1;
        const bool fullTile = (tile * 16 + 16 <= N);   // scalar

        // A fragments: lane holds row `lo`, K runs [c*32+kh8 .. +7] and [+16 .. +23]
        v16bf a[2];
#pragma unroll
        for (int c = 0; c < 2; ++c) {
            const float* p0 = x + (size_t)rowA * NF + c * 32 + kh8;
            const float* p1 = p0 + 16;
#pragma unroll
            for (int j = 0; j < 8; ++j) {
                a[c][j]     = (__bf16)p0[j];
                a[c][8 + j] = (__bf16)p1[j];
            }
        }

#pragma unroll
        for (int nt = 0; nt < 4; ++nt) {
            const int col = nt * 16 + lo;
            const float bias = bn[col];
            v8f acc = {bias, bias, bias, bias, bias, bias, bias, bias};
#pragma unroll
            for (int c = 0; c < 2; ++c) {
                const __bf16* bp = WnT + (size_t)col * NF + c * 32 + kh8;
                v8bf b0 = *(const v8bf*)bp;
                v8bf b1 = *(const v8bf*)(bp + 16);
                v16bf b = SHUF16(b0, b1);
                acc = __builtin_amdgcn_wmma_f32_16x16x32_bf16(
                          false, a[c], false, b, (short)0, acc, false, false);
            }
            __bf16* np = node_enc + (size_t)(tile * 16 + kh8) * NE + col;
            if (fullTile) {
#pragma unroll
                for (int r = 0; r < 8; ++r)
                    np[(size_t)r * NE] = (__bf16)lrelu(acc[r]);
            } else {
#pragma unroll
                for (int r = 0; r < 8; ++r) {
                    if (tile * 16 + r + kh8 < N)
                        np[(size_t)r * NE] = (__bf16)lrelu(acc[r]);
                }
            }
        }
    }
}

// ---------------------------------------------------------------------------
// Fused edge pipeline, one wave per 16-edge tile:
//   edge_enc^T = We^T(64x32) x edge_attr^T(32x16)  -> 4 WMMAs (bias in C)
//   feats A-frags assembled in registers (enc chunks + node_enc gather)
//   msg = lrelu(feats(16x128) x Wg(128x128) + bg)  -> 32 WMMAs (bias in C)
//   scatter: atomicAdd into out[dst][:]  (L2-resident)
// ---------------------------------------------------------------------------
__global__ void __launch_bounds__(256) edge_fused(
    const int* __restrict__ ei, const float* __restrict__ ea,
    const __bf16* __restrict__ WeTg, const __bf16* __restrict__ WgTg,
    const float* __restrict__ be, const float* __restrict__ bg,
    const __bf16* __restrict__ node_enc, float* __restrict__ out,
    int E, int eTiles)
{
    __shared__ __align__(16) __bf16 sWgT[OF * KF];  // 32 KB
    __shared__ __align__(16) __bf16 sWeT[EE * EF];  // 4 KB

    for (int i = threadIdx.x; i < (OF * KF) / 8; i += blockDim.x)
        ((v8bf*)sWgT)[i] = ((const v8bf*)WgTg)[i];
    for (int i = threadIdx.x; i < (EE * EF) / 8; i += blockDim.x)
        ((v8bf*)sWeT)[i] = ((const v8bf*)WeTg)[i];
    __syncthreads();

    const int lane = threadIdx.x & 31;
    const int wave = threadIdx.x >> 5;
    const int lo = lane & 15;
    const int hi = lane >> 4;
    const int kh8 = 8 * hi;

    // Preload We^T A-fragments (row = out-feature, K = 32 in-features)
    v16bf weA[4];
#pragma unroll
    for (int t = 0; t < 4; ++t) {
        const __bf16* p = sWeT + (t * 16 + lo) * EF;
        v8bf a0 = *(const v8bf*)(p + 16 * hi);
        v8bf a1 = *(const v8bf*)(p + 16 * hi + 8);
        weA[t] = SHUF16(a0, a1);
    }
    // Biases: be indexed by feature (row of enc^T frag), bg by output column
    float beF[32];
#pragma unroll
    for (int t = 0; t < 4; ++t)
#pragma unroll
        for (int r = 0; r < 8; ++r)
            beF[t * 8 + r] = be[t * 16 + kh8 + r];
    float bgF[8];
#pragma unroll
    for (int nt = 0; nt < 8; ++nt) bgF[nt] = bg[nt * 16 + lo];

    const int* srcIdx = ei;
    const int* dstIdx = ei + E;

    // Wave-uniform tile index in an SGPR -> scalar control flow.
    int tile = __builtin_amdgcn_readfirstlane(blockIdx.x * (blockDim.x >> 5) + wave);
    const int tstride = gridDim.x * (blockDim.x >> 5);
    for (; tile < eTiles; tile += tstride) {
        const int e0 = tile * 16;                      // scalar
        const bool fullTile = (e0 + 16 <= E);          // scalar
        int myE = e0 + lo;
        if (myE >= E) myE = E - 1;
        const int srcN = srcIdx[myE];
        const int dstBase = dstIdx[myE] * OF;

        // Prefetch next tile's edge_attr row (scalar guard, speculative).
        if (tile + tstride < eTiles)
            __builtin_prefetch(ea + ((size_t)myE + (size_t)tstride * 16) * EF, 0, 1);

        // Hoisted dst scatter offsets: dBase[r] = dst[e0 + r + kh8] * OF
        int dBase[8];
#pragma unroll
        for (int r = 0; r < 8; ++r)
            dBase[r] = __shfl(dstBase, r + kh8, 32);

        // B-frag of edge_attr^T: col = edge lo, K = 16*hi + 0..15 (contiguous row)
        v16bf eaB;
        {
            const float* p = ea + (size_t)myE * EF + 16 * hi;
#pragma unroll
            for (int j = 0; j < 16; ++j) eaB[j] = (__bf16)p[j];
        }

        // Edge encoder (transposed): 4 feature tiles of 16, bias preloaded in C.
        v8f ct[4];
#pragma unroll
        for (int t = 0; t < 4; ++t) {
            v8f z;
#pragma unroll
            for (int r = 0; r < 8; ++r) z[r] = beF[t * 8 + r];
            ct[t] = __builtin_amdgcn_wmma_f32_16x16x32_bf16(
                        false, weA[t], false, eaB, (short)0, z, false, false);
#pragma unroll
            for (int r = 0; r < 8; ++r)
                ct[t][r] = lrelu(ct[t][r]);
        }

        // Assemble feats A-fragments (16 edges x 128 K) in registers.
        v16bf aM[4];
#pragma unroll
        for (int j = 0; j < 8; ++j) {
            aM[0][j]     = (__bf16)ct[0][j];
            aM[0][8 + j] = (__bf16)ct[1][j];
            aM[1][j]     = (__bf16)ct[2][j];
            aM[1][8 + j] = (__bf16)ct[3][j];
        }
        {   // gather x_j = node_enc[src] (bf16, L2-resident)
            const __bf16* nrow = node_enc + (size_t)srcN * NE;
            v8bf g0 = *(const v8bf*)(nrow + kh8);
            v8bf g1 = *(const v8bf*)(nrow + 16 + kh8);
            v8bf g2 = *(const v8bf*)(nrow + 32 + kh8);
            v8bf g3 = *(const v8bf*)(nrow + 48 + kh8);
            aM[2] = SHUF16(g0, g1);
            aM[3] = SHUF16(g2, g3);
        }

        // Message GEMM: 8 N-tiles x 4 K-chunks (bias in C), lrelu + scatter.
#pragma unroll
        for (int nt = 0; nt < 8; ++nt) {
            const int col = nt * 16 + lo;
            const float bias = bgF[nt];
            v8f acc = {bias, bias, bias, bias, bias, bias, bias, bias};
#pragma unroll
            for (int c = 0; c < 4; ++c) {
                const __bf16* p = sWgT + (nt * 16 + lo) * KF + c * 32 + kh8;
                v8bf b0 = *(const v8bf*)p;
                v8bf b1 = *(const v8bf*)(p + 16);
                v16bf b = SHUF16(b0, b1);
                acc = __builtin_amdgcn_wmma_f32_16x16x32_bf16(
                          false, aM[c], false, b, (short)0, acc, false, false);
            }
            float* outCol = out + col;
            if (fullTile) {
#pragma unroll
                for (int r = 0; r < 8; ++r)
                    atomicAdd(outCol + dBase[r], lrelu(acc[r]));
            } else {
#pragma unroll
                for (int r = 0; r < 8; ++r) {
                    if (e0 + r + kh8 < E)
                        atomicAdd(outCol + dBase[r], lrelu(acc[r]));
                }
            }
        }
    }
}

// ---------------------------------------------------------------------------
extern "C" void kernel_launch(void* const* d_in, const int* in_sizes, int n_in,
                              void* d_out, int out_size, void* d_ws, size_t ws_size,
                              hipStream_t stream) {
    (void)n_in; (void)ws_size;
    const float* x   = (const float*)d_in[0];
    const int*   ei  = (const int*)d_in[1];
    const float* ea  = (const float*)d_in[2];
    const float* Wn  = (const float*)d_in[3];
    const float* bn  = (const float*)d_in[4];
    const float* We  = (const float*)d_in[5];
    const float* be  = (const float*)d_in[6];
    const float* Wg  = (const float*)d_in[7];
    const float* bg  = (const float*)d_in[8];
    float* out = (float*)d_out;

    const int N = in_sizes[0] / NF;
    const int E = in_sizes[2] / EF;

    // Workspace layout (256B aligned slices)
    char* ws = (char*)d_ws;
    size_t off = 0;
    __bf16* node_enc = (__bf16*)(ws + off);
    off += ((size_t)N * NE * sizeof(__bf16) + 255) & ~(size_t)255;
    __bf16* WnT = (__bf16*)(ws + off); off += ((size_t)NF * NE * 2 + 255) & ~(size_t)255;
    __bf16* WeT = (__bf16*)(ws + off); off += ((size_t)EF * EE * 2 + 255) & ~(size_t)255;
    __bf16* WgT = (__bf16*)(ws + off); off += ((size_t)KF * OF * 2 + 255) & ~(size_t)255;

    // 1) weight prep (tiny)
    prep_weights<<<64, 256, 0, stream>>>(Wn, We, Wg, WnT, WeT, WgT);

    // 2) node encoder
    const int nTiles = (N + 15) / 16;
    int nBlocks = (nTiles + 7) / 8;
    if (nBlocks > 4096) nBlocks = 4096;
    node_encoder<<<nBlocks, 256, 0, stream>>>(x, WnT, bn, node_enc, N, nTiles);

    // 3) zero output (graph-capturable memset node)
    hipMemsetAsync(d_out, 0, (size_t)out_size * sizeof(float), stream);

    // 4) fused edge pipeline
    const int eTiles = (E + 15) / 16;
    int eBlocks = (eTiles + 7) / 8;
    if (eBlocks > 1600) eBlocks = 1600;   // grid-stride; amortize 36KB LDS fill
    edge_fused<<<eBlocks, 256, 0, stream>>>(ei, ea, WeT, WgT, be, bg,
                                            node_enc, out, E, eTiles);
}
